// Attention_88201448391331
// MI455X (gfx1250) — matrix operand order repaired and checked by reading, not compiled, not run on hardware
//
#include <hip/hip_runtime.h>

typedef _Float16 v16h __attribute__((ext_vector_type(16)));
typedef _Float16 v8h  __attribute__((ext_vector_type(8)));
typedef _Float16 v2h  __attribute__((ext_vector_type(2)));
typedef float    v8f  __attribute__((ext_vector_type(8)));
typedef float    v4f  __attribute__((ext_vector_type(4)));
typedef int      v8i  __attribute__((ext_vector_type(8)));

#define SEQ      2048
#define DHEAD    64
#define BLOCK_M  128
#define BLOCK_N  32
#define NTHREADS 256
#define VSTRIDE  40                      // padded V^T row stride (halves); 80B = 5*16B
#define MASKVAL  (-1.0e30f)

// xor-16 half-wave swap via DS swizzle (group-of-32: xor=0x10, or=0, and=0x1f)
__device__ __forceinline__ int swap16i(int x) {
    return __builtin_amdgcn_ds_swizzle(x, 0x401f);
}
__device__ __forceinline__ float swap16f(float x) {
    return __builtin_bit_cast(float, swap16i(__builtin_bit_cast(int, x)));
}

__global__ __launch_bounds__(NTHREADS)
void fa_causal_wmma(const float* __restrict__ Qg,
                    const float* __restrict__ Kg,
                    const float* __restrict__ Vg,
                    float* __restrict__ Og) {
    __shared__ __align__(16) _Float16 Kl[BLOCK_N * DHEAD];   // [key][d] row-major
    __shared__ __align__(16) _Float16 Vt[DHEAD * VSTRIDE];   // [d][key] transposed, padded

    const int tid  = threadIdx.x;
    const int lane = tid & 31;
    const int wave = tid >> 5;
    const int col  = lane & 15;          // query column within tile / A-row index
    const bool hi  = lane >= 16;

    const int bh = blockIdx.x >> 4;                     // 0..63  (B*H)
    const int qb = (blockIdx.x & 15) * BLOCK_M;         // query block base
    const int qw = qb + wave * 16;                      // this wave's 16 query rows
    const int q  = qw + col;                            // this lane's query
    const size_t base = (size_t)bh * SEQ * DHEAD;

    // wave-uniform causal bounds, forced into SGPRs -> scalar branches around WMMA
    const int qw_s  = __builtin_amdgcn_readfirstlane(qw);
    const int jmax  = __builtin_amdgcn_readfirstlane((qw + 15) >> 5); // last block computed

    // ---- staging geometry: thread owns 8 contiguous floats of each 32x64 tile
    const int seg = tid * 8;             // element index within tile (0..2047)
    const int kr  = seg >> 6;            // key row this thread stages
    const int dbs = seg & 63;            // feature base this thread stages
    const float* kq = Kg + base + seg;   // advances by BLOCK_N*DHEAD per block
    const float* vq = Vg + base + seg;
    _Float16* klp = &Kl[seg];            // Kl is linear in seg
    const int vtb = dbs * VSTRIDE + kr;  // Vt scatter base

    // ---- Q fragments in B-format (B = Q^T, 64 x 16), pre-scaled by 1/sqrt(64)
    v16h bq[2];
    {
        const float* qrow = Qg + base + (size_t)q * DHEAD;
        #pragma unroll
        for (int s = 0; s < 2; ++s) {
            const int d0 = (hi ? 16 : 0) + 32 * s;
            #pragma unroll
            for (int e = 0; e < 16; ++e)
                bq[s][e] = (_Float16)(qrow[d0 + e] * 0.125f);
        }
    }

    // ---- O^T accumulator: 4 chunks of 16(d) x 16(q), C-format
    v8f acc[4];
    #pragma unroll
    for (int c = 0; c < 4; ++c)
        #pragma unroll
        for (int r = 0; r < 8; ++r) acc[c][r] = 0.0f;

    float m = MASKVAL;   // running row max
    float l = 0.0f;      // running row sum

    const int nkb = qb / BLOCK_N + (BLOCK_M / BLOCK_N);  // causal end for this q-block

    for (int jb = 0; jb < nkb; ++jb) {
        const int kb = jb * BLOCK_N;

        __syncthreads();   // previous iteration's LDS readers done

        // ---- cooperative stage: 4x global_load_b128 per thread, f32 -> f16
        v4f k0  = *(const v4f*)(kq);
        v4f k1  = *(const v4f*)(kq + 4);
        v4f vv0 = *(const v4f*)(vq);
        v4f vv1 = *(const v4f*)(vq + 4);
        kq += BLOCK_N * DHEAD;
        vq += BLOCK_N * DHEAD;
        if (jb + 1 < nkb) {              // prefetch next tile into cache
            __builtin_prefetch(kq, 0, 1);
            __builtin_prefetch(vq, 0, 1);
        }
        v8h hk;
        #pragma unroll
        for (int j = 0; j < 4; ++j) {
            hk[j]     = (_Float16)k0[j];
            hk[4 + j] = (_Float16)k1[j];
        }
        *(v8h*)klp = hk;                 // one ds_store_b128
        #pragma unroll
        for (int j = 0; j < 4; ++j) {
            Vt[vtb + j * VSTRIDE]       = (_Float16)vv0[j];
            Vt[vtb + (4 + j) * VSTRIDE] = (_Float16)vv1[j];
        }
        __syncthreads();

        if (jb > jmax) continue;   // SGPR compare: block entirely above diagonal

        // ---- S^T = K * Q^T : two 16x16 tiles (keys kb..kb+15, kb+16..kb+31)
        v8f ct[2];
        #pragma unroll
        for (int t = 0; t < 2; ++t) {
            v8f c;
            #pragma unroll
            for (int r = 0; r < 8; ++r) c[r] = 0.0f;
            #pragma unroll
            for (int s = 0; s < 2; ++s) {
                const int dlo = (hi ? 16 : 0) + 32 * s;
                const v8h* pa = (const v8h*)&Kl[(16 * t + col) * DHEAD + dlo];
                const v8h* pb = (const v8h*)&Kl[(16 * t + col) * DHEAD + dlo + 8];
                v16h a = __builtin_shufflevector(*pa, *pb,
                        0,1,2,3,4,5,6,7,8,9,10,11,12,13,14,15);
                c = __builtin_amdgcn_wmma_f32_16x16x32_f16(
                        false, a, false, bq[s], (short)0, c, false, false);
            }
            ct[t] = c;
        }

        // ---- causal mask (C-format of S^T: key in register, query in lane)
        if (kb + BLOCK_N - 1 > qw_s) {   // SGPR compare; per-lane mask below is VALU-only
            #pragma unroll
            for (int t = 0; t < 2; ++t)
                #pragma unroll
                for (int r = 0; r < 8; ++r) {
                    const int key = kb + 16 * t + r + (hi ? 8 : 0);
                    if (key > q) ct[t][r] = MASKVAL;
                }
        }

        // ---- online softmax over keys (in-register + one xor-16 swizzle)
        float mb = MASKVAL;
        #pragma unroll
        for (int t = 0; t < 2; ++t)
            #pragma unroll
            for (int r = 0; r < 8; ++r) mb = fmaxf(mb, ct[t][r]);
        mb = fmaxf(mb, swap16f(mb));
        const float mnew = fmaxf(m, mb);

        float p0f[8], p1f[8];
        float lb = 0.0f;
        #pragma unroll
        for (int r = 0; r < 8; ++r) { p0f[r] = __expf(ct[0][r] - mnew); lb += p0f[r]; }
        #pragma unroll
        for (int r = 0; r < 8; ++r) { p1f[r] = __expf(ct[1][r] - mnew); lb += p1f[r]; }
        lb += swap16f(lb);

        const float scale = __expf(m - mnew);
        m = mnew;
        l = l * scale + lb;
        #pragma unroll
        for (int c = 0; c < 4; ++c)
            #pragma unroll
            for (int r = 0; r < 8; ++r) acc[c][r] *= scale;

        // ---- P^T in B-format (32 keys x 16 queries): half-wave swap, 8 swizzles
        v8i w;
        #pragma unroll
        for (int j = 0; j < 4; ++j) {
            v2h o0 = { (_Float16)p0f[2 * j], (_Float16)p0f[2 * j + 1] };
            v2h o1 = { (_Float16)p1f[2 * j], (_Float16)p1f[2 * j + 1] };
            const int i0 = __builtin_bit_cast(int, o0);
            const int i1 = __builtin_bit_cast(int, o1);
            const int s0 = swap16i(i0);
            const int s1 = swap16i(i1);
            w[j]     = hi ? s1 : i0;   // elements 0..7  : local keys klo+0..7
            w[4 + j] = hi ? i1 : s0;   // elements 8..15 : local keys klo+8..15
        }
        const v16h bp = __builtin_bit_cast(v16h, w);

        // ---- O^T += V^T * P^T   (4 d-chunks, K = 32 keys)
        #pragma unroll
        for (int c = 0; c < 4; ++c) {
            const int klo = hi ? 16 : 0;
            const v8h* pa = (const v8h*)&Vt[(16 * c + col) * VSTRIDE + klo];
            const v8h* pb = (const v8h*)&Vt[(16 * c + col) * VSTRIDE + klo + 8];
            v16h a = __builtin_shufflevector(*pa, *pb,
                    0,1,2,3,4,5,6,7,8,9,10,11,12,13,14,15);
            acc[c] = __builtin_amdgcn_wmma_f32_16x16x32_f16(
                    false, a, false, bp, (short)0, acc[c], false, false);
        }
    }

    // ---- epilogue: O[q][d] = acc / l
    const float inv = 1.0f / l;
    #pragma unroll
    for (int c = 0; c < 4; ++c)
        #pragma unroll
        for (int r = 0; r < 8; ++r) {
            const int d = 16 * c + r + (hi ? 8 : 0);
            Og[base + (size_t)q * DHEAD + d] = acc[c][r] * inv;
        }
}

extern "C" void kernel_launch(void* const* d_in, const int* in_sizes, int n_in,
                              void* d_out, int out_size, void* d_ws, size_t ws_size,
                              hipStream_t stream) {
    const float* Q = (const float*)d_in[0];
    const float* K = (const float*)d_in[1];
    const float* V = (const float*)d_in[2];
    float* O = (float*)d_out;
    // grid: (B*H) * (SEQ / BLOCK_M) = 64 * 16
    dim3 grid(64 * (SEQ / BLOCK_M)), block(NTHREADS);
    fa_causal_wmma<<<grid, block, 0, stream>>>(Q, K, V, O);
}
